// KernelDenseBayesian_10024453669624
// MI455X (gfx1250) — matrix operand order repaired and checked by reading, not compiled, not run on hardware
//
#include <hip/hip_runtime.h>

#define B_SZ   8192
#define IN_SZ  4096
#define OUT_SZ 4096

typedef __bf16 bf16;
typedef __attribute__((ext_vector_type(16))) __bf16 v16bf;
typedef __attribute__((ext_vector_type(8)))  __bf16 v8bf;
typedef __attribute__((ext_vector_type(4)))  __bf16 v4bf;
typedef __attribute__((ext_vector_type(8)))  float  v8f;
typedef __attribute__((ext_vector_type(4)))  unsigned int u32x4;
typedef __attribute__((ext_vector_type(8)))  int i32x8;
typedef __attribute__((ext_vector_type(4)))  int i32x4;

// LDS tile geometry: 128 rows x 32 bf16, padded by TDM to a 40-element row
// stride (pad 4 dwords after every 16 dwords) -> 20-bank stride, conflict-free.
#define TILE_ROW_ELEMS   40
#define TILE_ELEMS       (128 * TILE_ROW_ELEMS)   // 5120 elems = 10240 B
#define BUF_ELEMS        (2 * TILE_ELEMS)         // A tile + B tile = 20480 B
#define BUF_BYTES        (BUF_ELEMS * 2)

// ---------------------------------------------------------------------------
// Kernel 1: wT[o][i] = bf16( exp( -|| col_i - row_o ||^2 ) ),  [OUT, IN] K-major
// ---------------------------------------------------------------------------
__global__ __launch_bounds__(256) void build_wt_kernel(
    const float* __restrict__ rows_mean,   // [OUT, 2]
    const float* __restrict__ cols_mean,   // [IN, 2]
    bf16* __restrict__ wt)                 // [OUT, IN]
{
    unsigned idx = blockIdx.x * 256u + threadIdx.x;   // < 16M
    unsigned o = idx >> 12;                           // / IN  (IN = 4096)
    unsigned i = idx & 4095u;                         // % IN
    float r0 = rows_mean[2u * o + 0], r1 = rows_mean[2u * o + 1];
    float c0 = cols_mean[2u * i + 0], c1 = cols_mean[2u * i + 1];
    float d0 = c0 - r0, d1 = c1 - r1;
    wt[idx] = (bf16)__expf(-(d0 * d0 + d1 * d1));
}

// ---------------------------------------------------------------------------
// Kernel 2: xa[b][k] = bf16( x[b][k] * alpha[k] ),  [B, IN]
// ---------------------------------------------------------------------------
__global__ __launch_bounds__(256) void scale_x_kernel(
    const float* __restrict__ x,
    const float* __restrict__ alpha,
    bf16* __restrict__ xa)
{
    size_t idx4 = (size_t)(blockIdx.x * 256u + threadIdx.x) * 4u;
    unsigned k = (unsigned)(idx4 & 4095u);
    float4 xv = *(const float4*)(x + idx4);
    float4 av = *(const float4*)(alpha + k);
    v4bf o;
    o[0] = (bf16)(xv.x * av.x);
    o[1] = (bf16)(xv.y * av.y);
    o[2] = (bf16)(xv.z * av.z);
    o[3] = (bf16)(xv.w * av.w);
    *(v4bf*)(xa + idx4) = o;
}

// ---------------------------------------------------------------------------
// TDM: issue one 2D tile load (128 rows x 32 bf16) global -> LDS.
// ---------------------------------------------------------------------------
__device__ __forceinline__ void tdm_load_tile(const bf16* gptr, unsigned lds_off,
                                              unsigned tensor_rows)
{
    unsigned long long ga = (unsigned long long)(size_t)gptr;
    u32x4 g0;
    g0[0] = 1u;                                            // count=1, user D#
    g0[1] = lds_off;                                       // LDS byte address
    g0[2] = (unsigned)(ga & 0xFFFFFFFFu);                  // global_addr[31:0]
    g0[3] = (unsigned)((ga >> 32) & 0x1FFFFFFu) | (2u << 30); // addr[56:32] | type=2
    i32x8 g1;
    g1[0] = (1 << 16)   // data_size = 2 bytes
          | (1 << 20)   // pad_enable
          | (3 << 22)   // pad_interval code 3 = 16 dwords (one 32-elem row)
          | (3 << 25);  // pad_amount  code 3 = 4 dwords  (8 elems)
    g1[1] = (int)((IN_SZ & 0xFFFFu) << 16);                         // tensor_dim0 lo16
    g1[2] = (int)((IN_SZ >> 16) | ((tensor_rows & 0xFFFFu) << 16)); // td0 hi | td1 lo
    g1[3] = (int)((tensor_rows >> 16) | (32u << 16));               // td1 hi | tile_dim0=32
    g1[4] = 128;                                                    // tile_dim1=128
    g1[5] = IN_SZ;                                                  // tensor_dim0_stride lo32
    g1[6] = 0;
    g1[7] = 0;
    i32x4 z4 = {0, 0, 0, 0};
    i32x8 z8 = {0, 0, 0, 0, 0, 0, 0, 0};
    __builtin_amdgcn_tensor_load_to_lds(g0, g1, z4, z4, z8, 0);
}

// ---------------------------------------------------------------------------
// One K=32 compute step: read 2 A frags + 4 B frags from an LDS buffer and
// run 8 WMMAs. The LDS pointers are passed into empty inline asm as
// INPUT-ONLY operands with a memory clobber: this escapes the TDM-written
// tiles (so the loads can't be undef-folded) while keeping the actual load
// address chain rooted at `smem`, so address-space inference still emits
// ds_load_b128 rather than flat loads.
// ---------------------------------------------------------------------------
__device__ __forceinline__ void compute_step(
    const bf16* As, const bf16* Bs,
    int wave_m, int wave_n, int half, int l16,
    v8f (&acc)[2][4])
{
    asm volatile("" : : "v"((const void*)As), "v"((const void*)Bs) : "memory");

    v16bf a[2];
#pragma unroll
    for (int i = 0; i < 2; ++i) {
        int row = wave_m * 32 + i * 16 + l16;
        const bf16* pa = As + row * TILE_ROW_ELEMS + half * 8;
        v8bf lo = *(const v8bf*)(pa);        // K = 8h .. 8h+7
        v8bf hi = *(const v8bf*)(pa + 16);   // K = 16+8h .. 16+8h+7
#pragma unroll
        for (int e = 0; e < 8; ++e) { a[i][e] = lo[e]; a[i][8 + e] = hi[e]; }
    }
    v16bf b[4];
#pragma unroll
    for (int j = 0; j < 4; ++j) {
        int row = wave_n * 64 + j * 16 + l16;
        const bf16* pb = Bs + row * TILE_ROW_ELEMS + half * 8;
        v8bf lo = *(const v8bf*)(pb);        // K = 8h .. 8h+7
        v8bf hi = *(const v8bf*)(pb + 16);   // K = 16+8h .. 16+8h+7
#pragma unroll
        for (int e = 0; e < 8; ++e) { b[j][e] = lo[e]; b[j][8 + e] = hi[e]; }
    }

#pragma unroll
    for (int i = 0; i < 2; ++i)
#pragma unroll
        for (int j = 0; j < 4; ++j)
            acc[i][j] = __builtin_amdgcn_wmma_f32_16x16x32_bf16(
                false, a[i], false, b[j], (short)0, acc[i][j], false, false);
}

// ---------------------------------------------------------------------------
// Kernel 3: C[M,N] = A[M,K] * Wt[N,K]^T  (bf16 WMMA, f32 acc)
// 256 threads = 8 waves; block tile 128x128; wave tile 32x64 (2x4 frags).
// TDM double-buffers A/B tiles in LDS; K-loop unrolled by 2 so the buffer
// parity (and hence every LDS offset in the descriptors) is static.
// ---------------------------------------------------------------------------
__global__ __launch_bounds__(256) void gemm_kernel(
    const bf16* __restrict__ A,    // [B_SZ, IN_SZ] row-major
    const bf16* __restrict__ Wt,   // [OUT_SZ, IN_SZ] row-major
    float* __restrict__ C)         // [B_SZ, OUT_SZ] row-major
{
    // Single __shared__ allocation -> LDS offset 0 for the TDM descriptors.
    __shared__ __align__(16) bf16 smem[2 * BUF_ELEMS];   // 40 KB

    const int K = IN_SZ;
    const int wave   = threadIdx.x >> 5;
    const int lane   = threadIdx.x & 31;
    const int half   = lane >> 4;
    const int l16    = lane & 15;
    const int wave_m = wave & 3;      // 4 waves along M -> 32 rows each
    const int wave_n = wave >> 2;     // 2 waves along N -> 64 cols each

    const unsigned rowb = blockIdx.y * 128u;
    const unsigned colb = blockIdx.x * 128u;
    const bf16* Ablk = A  + (size_t)rowb * K;
    const bf16* Bblk = Wt + (size_t)colb * K;

    const bf16* As0 = smem;
    const bf16* Bs0 = smem + TILE_ELEMS;
    const bf16* As1 = smem + BUF_ELEMS;
    const bf16* Bs1 = smem + BUF_ELEMS + TILE_ELEMS;

    v8f acc[2][4];
#pragma unroll
    for (int i = 0; i < 2; ++i)
#pragma unroll
        for (int j = 0; j < 4; ++j)
#pragma unroll
            for (int v = 0; v < 8; ++v) acc[i][j][v] = 0.0f;

    // Prologue: stage k=0 into buffer 0.
    if (wave == 0) {
        tdm_load_tile(Ablk, 0u, B_SZ - rowb);
        tdm_load_tile(Bblk, TILE_ELEMS * 2u, OUT_SZ - colb);
    }
    __builtin_amdgcn_s_wait_tensorcnt(0);
    __syncthreads();

    for (int k = 0; k < K; k += 64) {
        // Prefetch k+32 into buffer 1 (always in range: k+32 <= K-32).
        if (wave == 0) {
            tdm_load_tile(Ablk + (k + 32), BUF_BYTES, B_SZ - rowb);
            tdm_load_tile(Bblk + (k + 32), BUF_BYTES + TILE_ELEMS * 2u, OUT_SZ - colb);
        }
        compute_step(As0, Bs0, wave_m, wave_n, half, l16, acc);
        __builtin_amdgcn_s_wait_tensorcnt(0);
        __syncthreads();

        // Prefetch k+64 into buffer 0 (skip on the final pass).
        if (wave == 0 && (k + 64) < K) {
            tdm_load_tile(Ablk + (k + 64), 0u, B_SZ - rowb);
            tdm_load_tile(Bblk + (k + 64), TILE_ELEMS * 2u, OUT_SZ - colb);
        }
        compute_step(As1, Bs1, wave_m, wave_n, half, l16, acc);
        __builtin_amdgcn_s_wait_tensorcnt(0);
        __syncthreads();
    }

    // C/D layout: lane(h, n), VGPR v -> C[row_tile + v + 8h][col_tile + n]
#pragma unroll
    for (int i = 0; i < 2; ++i)
#pragma unroll
        for (int j = 0; j < 4; ++j) {
            float* cp = C + ((size_t)rowb + wave_m * 32 + i * 16 + half * 8) * OUT_SZ
                          + colb + wave_n * 64 + j * 16 + l16;
#pragma unroll
            for (int v = 0; v < 8; ++v)
                cp[(size_t)v * OUT_SZ] = acc[i][j][v];
        }
}

// ---------------------------------------------------------------------------
extern "C" void kernel_launch(void* const* d_in, const int* in_sizes, int n_in,
                              void* d_out, int out_size, void* d_ws, size_t ws_size,
                              hipStream_t stream) {
    const float* x         = (const float*)d_in[0];  // [8192, 4096]
    const float* rows_mean = (const float*)d_in[1];  // [4096, 2]
    const float* cols_mean = (const float*)d_in[2];  // [4096, 2]
    const float* alpha     = (const float*)d_in[3];  // [4096]
    float* out = (float*)d_out;                      // [8192, 4096]

    bf16* xa = (bf16*)d_ws;                                                // 64 MB
    bf16* wt = (bf16*)((char*)d_ws + (size_t)B_SZ * IN_SZ * sizeof(bf16)); // +32 MB

    build_wt_kernel<<<(OUT_SZ * IN_SZ) / 256, 256, 0, stream>>>(rows_mean, cols_mean, wt);
    scale_x_kernel<<<(B_SZ * IN_SZ) / (4 * 256), 256, 0, stream>>>(x, alpha, xa);

    dim3 grid(OUT_SZ / 128, B_SZ / 128);
    gemm_kernel<<<grid, 256, 0, stream>>>(xa, wt, out);
}
